// LayerNormGRU_39745627357261
// MI455X (gfx1250) — matrix operand order repaired and checked by reading, not compiled, not run on hardware
//
#include <hip/hip_runtime.h>
#include <math.h>

// ---------------------------------------------------------------------------
// LayerNorm-GRU, B=32, T=1024, IN=H=512, L=2   (MI455X / gfx1250, wave32)
//
//  (1) convert x, Wi, Wh to bf16 in workspace
//  (2) k_xgemm: XG0[B*T,3H] = x_bf @ Wi0_bf^T   (parallel, off critical path)
//  (3) k_gru:  persistent 16-WG kernel scanning t=0..1023, l=0..1.
//      Each WG owns 32 H-columns (and matching r/z/n weight rows).
//      Per step/layer: bf16 WMMA GEMM slices (weights streamed from the
//      192MB-L2-resident bf16 copies), xg0 slice for step t+1 prefetched
//      into LDS with GLOBAL_LOAD_ASYNC_TO_LDS_B128 (double-buffered),
//      LN stats reduced across WGs via global partials + software grid
//      barrier, gates + state update pointwise.
// ---------------------------------------------------------------------------

#define Bsz   32
#define Tlen  1024
#define Hdim  512
#define H3    1536
#define NWG   16
#define COLS  32          // H-columns per workgroup (16*32 = 512)
#define EPSLN 1e-5f

typedef __bf16 v16bf __attribute__((ext_vector_type(16)));
typedef __bf16 v8bf  __attribute__((ext_vector_type(8)));
typedef float  v8f   __attribute__((ext_vector_type(8)));

// -------- workspace layout (bytes) --------
#define OFF_CTRL   ((size_t)0)            // 2 x u32 (barrier count, generation)
#define OFF_SRZ    ((size_t)1024)         // [NWG][32][4] f32 partial r/z stats
#define OFF_SN     ((size_t)10240)        // [NWG][32][2] f32 partial n stats
#define OFF_HF     ((size_t)16384)        // [2][32][512] f32 hidden state
#define OFF_HBF    ((size_t)147456)       // [2][32][512] bf16 hidden state
#define OFF_WBF    ((size_t)262144)       // [Wi0,Wi1,Wh0,Wh1] bf16, 786432 ea
#define OFF_XBF    ((size_t)8388608)      // [B*T,512] bf16 x
#define OFF_XG0    ((size_t)50331648)     // [B*T,1536] f32 precomputed x-gates

__device__ __forceinline__ float sigmoidf(float x) {
  return 1.0f / (1.0f + __expf(-x));
}

// Async 16B global -> LDS copy (gfx1250 GLOBAL_LOAD_ASYNC_TO_LDS_B128, GV
// mode; VDST VGPR = wave-relative LDS byte offset, tracked by ASYNCcnt).
__device__ __forceinline__ void async_g2l_b128(unsigned lds_off,
                                               const void* gaddr) {
  asm volatile("global_load_async_to_lds_b128 %0, %1, off"
               :
               : "v"(lds_off), "v"(gaddr)
               : "memory");
}
__device__ __forceinline__ void wait_asynccnt0() {
  asm volatile("s_wait_asynccnt 0x0" ::: "memory");
}

// WMMA A-fragment (16x32 bf16): lane<16 holds K {kb..kb+7, kb+16..kb+23},
// lane>=16 holds K {kb+8..kb+15, kb+24..kb+31}; row M = lane&15.
__device__ __forceinline__ v16bf load_a_frag(const __bf16* __restrict__ row,
                                             int kb, int hi) {
  v8bf a0 = *(const v8bf*)(row + kb + 8 * hi);
  v8bf a1 = *(const v8bf*)(row + kb + 16 + 8 * hi);
  v16bf a;
#pragma unroll
  for (int i = 0; i < 8; ++i) { a[i] = a0[i]; a[i + 8] = a1[i]; }
  return a;
}

// B-fragment (32x16 bf16): lane's N = lane&15 = weight row; lane<16 holds
// K {kb..kb+7, kb+16..kb+23}, lane>=16 holds K {kb+8..kb+15, kb+24..kb+31}.
__device__ __forceinline__ v16bf load_b_frag(const __bf16* __restrict__ wrow,
                                             int kb, int hi) {
  v8bf b0 = *(const v8bf*)(wrow + kb + 8 * hi);
  v8bf b1 = *(const v8bf*)(wrow + kb + 16 + 8 * hi);
  v16bf b;
#pragma unroll
  for (int i = 0; i < 8; ++i) { b[i] = b0[i]; b[i + 8] = b1[i]; }
  return b;
}

// ---------------------------------------------------------------------------
__global__ void k_cvt_bf16(const float* __restrict__ src,
                           __bf16* __restrict__ dst, int n) {
  int i = blockIdx.x * blockDim.x + threadIdx.x;
  if (i < n) dst[i] = (__bf16)src[i];
}

__global__ void k_init(const float* __restrict__ h0, float* __restrict__ hf,
                       __bf16* __restrict__ hbf, unsigned* __restrict__ ctrl) {
  int i = blockIdx.x * blockDim.x + threadIdx.x;
  if (i < 2) ctrl[i] = 0u;
  if (i < 2 * Bsz * Hdim) {
    float v = h0[i];
    hf[i] = v;
    hbf[i] = (__bf16)v;
  }
}

// XG0 = x_bf[32768,512] @ Wi0_bf^T[512,1536] -> f32 [32768,1536]
// grid.x = 2048 m-tiles, 256 threads = 8 waves; wave w handles n-tiles
// w, w+8, ... (12 per wave). K loop = 16 x k32 WMMA steps.
__global__ __launch_bounds__(256) void k_xgemm(
    const __bf16* __restrict__ xbf, const __bf16* __restrict__ wi0,
    float* __restrict__ xg) {
  const int wave = threadIdx.x >> 5;
  const int lane = threadIdx.x & 31;
  const int hi = lane >> 4;
  const int ln = lane & 15;
  const int mt = blockIdx.x;
  const __bf16* arow = xbf + (size_t)(mt * 16 + ln) * Hdim;
  for (int nt = wave; nt < 96; nt += 8) {
    const __bf16* brow = wi0 + (size_t)(nt * 16 + ln) * Hdim;
    if (nt + 8 < 96)   // warm L0 for next weight tile (global_prefetch_b8)
      __builtin_prefetch(wi0 + (size_t)((nt + 8) * 16 + ln) * Hdim, 0, 3);
    v8f c = {};
#pragma unroll 4
    for (int kk = 0; kk < 16; ++kk) {
      int kb = kk * 32;
      v16bf a = load_a_frag(arow, kb, hi);
      v16bf b = load_b_frag(brow, kb, hi);
      c = __builtin_amdgcn_wmma_f32_16x16x32_bf16(false, a, false, b,
                                                  (short)0, c, false, false);
    }
#pragma unroll
    for (int v = 0; v < 8; ++v) {
      int m = mt * 16 + v + 8 * hi;
      int n = nt * 16 + ln;
      xg[(size_t)m * H3 + n] = c[v];
    }
  }
}

// -------- grid-wide sense barrier over NWG resident workgroups --------
__device__ __forceinline__ void gbar(unsigned* ctrl) {
  __syncthreads();
  if (threadIdx.x == 0) {
    __threadfence();
    volatile unsigned* gen = ctrl + 1;
    unsigned g = *gen;
    if (atomicAdd(&ctrl[0], 1u) == NWG - 1) {
      atomicExch(&ctrl[0], 0u);
      __threadfence();
      atomicAdd(&ctrl[1], 1u);
    } else {
      while (*gen == g) __builtin_amdgcn_s_sleep(2);
    }
    __threadfence();
  }
  __syncthreads();
}

// ---------------------------------------------------------------------------
// Persistent recurrent kernel: 16 blocks x 128 threads (4 waves).
// WG g owns H-columns [g*32, g*32+32) of every gate.
// ---------------------------------------------------------------------------
__global__ __launch_bounds__(128, 1) void k_gru(
    const float* __restrict__ xg0,                    // [B*T, 3H]
    const __bf16* __restrict__ wi1,                   // [3H, H] bf16
    const __bf16* __restrict__ wh0,                   // [3H, H] bf16
    const __bf16* __restrict__ wh1,                   // [3H, H] bf16
    const float* __restrict__ gr, const float* __restrict__ br,
    const float* __restrict__ gz, const float* __restrict__ bz,
    const float* __restrict__ gn, const float* __restrict__ bn,  // [L,H]
    float* __restrict__ hf,                           // [2][32][512] f32
    __bf16* __restrict__ hbf,                         // [2][32][512] bf16
    float* __restrict__ sRZ,                          // [NWG][32][4]
    float* __restrict__ sN,                           // [NWG][32][2]
    unsigned* __restrict__ ctrl,
    float* __restrict__ out,                          // [B,T,H]
    float* __restrict__ hn_out) {                     // [L,B,H]
  __shared__ float pre[3][Bsz][COLS];   // r/z preacts, hn
  __shared__ float xnb[Bsz][COLS];      // x contribution of n gate, then v_n
  __shared__ float red[Bsz][4][2];      // intra-WG n-stat reduce
  __shared__ __align__(16) float xgbuf[2][3][Bsz][COLS];  // async xg0 dbl-buf

  const int wg   = blockIdx.x;
  const int tid  = threadIdx.x;
  const int wave = tid >> 5;
  const int lane = tid & 31;
  const int hi   = lane >> 4;
  const int ln   = lane & 15;
  const int pb   = tid >> 2;            // pointwise: batch row 0..31
  const int pc0  = (tid & 3) * 8;       // pointwise: first of 8 columns

  // Issue this WG's xg0 slice (3 gates x 32 rows x 32 cols = 768 x 16B)
  // for timestep tt into LDS buffer `buf` via async-to-LDS DMA.
  auto issue_xg_prefetch = [&](int tt, int buf) {
    for (int q = tid; q < 768; q += 128) {
      const int gate = q >> 8;
      const int b = (q >> 3) & 31;
      const int sub = q & 7;
      const float* g = xg0 + ((size_t)b * Tlen + tt) * H3 + gate * Hdim +
                       wg * COLS + sub * 4;
      unsigned l = (unsigned)(size_t)(const void*)&xgbuf[buf][gate][b][sub * 4];
      async_g2l_b128(l, g);
    }
  };

  issue_xg_prefetch(0, 0);

  for (int t = 0; t < Tlen; ++t) {
    // xg0 slice for this step must be in LDS; kick off next step's DMA.
    wait_asynccnt0();
    __syncthreads();
    if (t + 1 < Tlen) issue_xg_prefetch(t + 1, (t + 1) & 1);
    const int xbuf = t & 1;

    for (int l = 0; l < 2; ++l) {
      const __bf16* Wh = l ? wh1 : wh0;
      const __bf16* Ah = hbf + (size_t)l * Bsz * Hdim;  // recurrent input
      const __bf16* Ax = hbf;                           // layer-1 input = h0

      // ---- GEMM phase: 12 (gate,mt,nt) tiles over 4 waves ----
      for (int c = wave * 3; c < wave * 3 + 3; ++c) {
        const int gate = c >> 2;
        const int mt = (c >> 1) & 1;
        const int nt = c & 1;
        const int wrow = gate * Hdim + wg * COLS + nt * 16 + ln;
        const __bf16* ahrow = Ah + (size_t)(mt * 16 + ln) * Hdim;
        const __bf16* bhrow = Wh + (size_t)wrow * Hdim;
        const __bf16* axrow = Ax + (size_t)(mt * 16 + ln) * Hdim;
        const __bf16* bxrow = wi1 + (size_t)wrow * Hdim;
        v8f ch = {};
        v8f cx = {};
#pragma unroll 2
        for (int kk = 0; kk < 16; ++kk) {
          int kb = kk * 32;
          v16bf a = load_a_frag(ahrow, kb, hi);
          v16bf b = load_b_frag(bhrow, kb, hi);
          ch = __builtin_amdgcn_wmma_f32_16x16x32_bf16(false, a, false, b,
                                                       (short)0, ch, false,
                                                       false);
          if (l == 1) {
            v16bf a2 = load_a_frag(axrow, kb, hi);
            v16bf b2 = load_b_frag(bxrow, kb, hi);
            cx = __builtin_amdgcn_wmma_f32_16x16x32_bf16(false, a2, false, b2,
                                                         (short)0, cx, false,
                                                         false);
          }
        }
        // writeback: C layout -> M = v + 8*hi, N = lane&15
#pragma unroll
        for (int v = 0; v < 8; ++v) {
          const int b = mt * 16 + v + 8 * hi;
          const int cl = nt * 16 + ln;                  // local column
          float xv;
          if (l == 0)
            xv = xgbuf[xbuf][gate][b][cl];              // async-prefetched
          else
            xv = cx[v];
          if (gate < 2)
            pre[gate][b][cl] = ch[v] + xv;
          else {
            pre[2][b][cl] = ch[v];                      // hn
            xnb[b][cl] = xv;                            // xn
          }
        }
      }
      __syncthreads();

      // ---- partial LN stats for r and z (per batch row, this WG's cols) ----
      if (tid < Bsz) {
        float sr = 0.f, qr = 0.f, sz = 0.f, qz = 0.f;
        for (int cc = 0; cc < COLS; ++cc) {
          float vr = pre[0][tid][cc]; sr += vr; qr += vr * vr;
          float vz = pre[1][tid][cc]; sz += vz; qz += vz * vz;
        }
        float* p = sRZ + ((size_t)wg * Bsz + tid) * 4;
        p[0] = sr; p[1] = qr; p[2] = sz; p[3] = qz;
      }
      gbar(ctrl);

      // ---- finalize r,z; build v_n = xn + r*hn; partial n stats ----
      {
        float sr = 0.f, qr = 0.f, sz = 0.f, qz = 0.f;
        for (int w = 0; w < NWG; ++w) {
          const float* p = sRZ + ((size_t)w * Bsz + pb) * 4;
          sr += p[0]; qr += p[1]; sz += p[2]; qz += p[3];
        }
        const float inv = 1.0f / (float)Hdim;
        float mr = sr * inv, vr = qr * inv - mr * mr;
        float mz = sz * inv, vz = qz * inv - mz * mz;
        float rsr = rsqrtf(vr + EPSLN), rsz = rsqrtf(vz + EPSLN);
        float sn = 0.f, qn = 0.f;
#pragma unroll
        for (int i = 0; i < 8; ++i) {
          int cl = pc0 + i;
          int hcol = wg * COLS + cl;
          float r = sigmoidf((pre[0][pb][cl] - mr) * rsr * gr[l * Hdim + hcol] +
                             br[l * Hdim + hcol]);
          float z = sigmoidf((pre[1][pb][cl] - mz) * rsz * gz[l * Hdim + hcol] +
                             bz[l * Hdim + hcol]);
          float vn = xnb[pb][cl] + r * pre[2][pb][cl];
          pre[1][pb][cl] = z;
          xnb[pb][cl] = vn;
          sn += vn; qn += vn * vn;
        }
        red[pb][tid & 3][0] = sn;
        red[pb][tid & 3][1] = qn;
      }
      __syncthreads();
      if (tid < Bsz) {
        float s = 0.f, q = 0.f;
        for (int j = 0; j < 4; ++j) { s += red[tid][j][0]; q += red[tid][j][1]; }
        float* p = sN + ((size_t)wg * Bsz + tid) * 2;
        p[0] = s; p[1] = q;
      }
      gbar(ctrl);

      // ---- finalize n, update hidden state ----
      {
        float sn = 0.f, qn = 0.f;
        for (int w = 0; w < NWG; ++w) {
          const float* p = sN + ((size_t)w * Bsz + pb) * 2;
          sn += p[0]; qn += p[1];
        }
        const float inv = 1.0f / (float)Hdim;
        float mn = sn * inv, vn = qn * inv - mn * mn;
        float rsn = rsqrtf(vn + EPSLN);
#pragma unroll
        for (int i = 0; i < 8; ++i) {
          int cl = pc0 + i;
          int hcol = wg * COLS + cl;
          float n = tanhf((xnb[pb][cl] - mn) * rsn * gn[l * Hdim + hcol] +
                          bn[l * Hdim + hcol]);
          float z = pre[1][pb][cl];
          size_t hidx = ((size_t)l * Bsz + pb) * Hdim + hcol;
          float hprev = hf[hidx];
          float hnew = (1.0f - z) * n + z * hprev;
          if (l == 1) hnew += hf[(size_t)pb * Hdim + hcol];   // residual: +h0_new
          hf[hidx] = hnew;
          hbf[hidx] = (__bf16)hnew;
          if (l == 1) out[((size_t)pb * Tlen + t) * Hdim + hcol] = hnew;
          if (t == Tlen - 1)
            hn_out[((size_t)l * Bsz + pb) * Hdim + hcol] = hnew;
        }
      }
      gbar(ctrl);
    }
  }
}

// ---------------------------------------------------------------------------
extern "C" void kernel_launch(void* const* d_in, const int* in_sizes, int n_in,
                              void* d_out, int out_size, void* d_ws,
                              size_t ws_size, hipStream_t stream) {
  (void)in_sizes; (void)n_in; (void)out_size; (void)ws_size;
  const float* x  = (const float*)d_in[0];   // [B,T,IN]
  const float* h0 = (const float*)d_in[1];   // [L,B,H]
  const float* Wi = (const float*)d_in[2];   // [L,3H,H]
  const float* Wh = (const float*)d_in[3];   // [L,3H,H]
  const float* gr = (const float*)d_in[4];
  const float* br = (const float*)d_in[5];
  const float* gz = (const float*)d_in[6];
  const float* bz = (const float*)d_in[7];
  const float* gn = (const float*)d_in[8];
  const float* bn = (const float*)d_in[9];

  char* ws = (char*)d_ws;
  unsigned* ctrl = (unsigned*)(ws + OFF_CTRL);
  float* sRZ = (float*)(ws + OFF_SRZ);
  float* sN  = (float*)(ws + OFF_SN);
  float* hf  = (float*)(ws + OFF_HF);
  __bf16* hbf = (__bf16*)(ws + OFF_HBF);
  __bf16* wbf = (__bf16*)(ws + OFF_WBF);
  __bf16* wi0 = wbf;
  __bf16* wi1 = wbf + (size_t)H3 * Hdim;
  __bf16* wh0 = wbf + (size_t)2 * H3 * Hdim;
  __bf16* wh1 = wbf + (size_t)3 * H3 * Hdim;
  __bf16* xbf = (__bf16*)(ws + OFF_XBF);
  float* xg0 = (float*)(ws + OFF_XG0);
  float* out = (float*)d_out;
  float* hn_out = out + (size_t)Bsz * Tlen * Hdim;

  const int nX = Bsz * Tlen * Hdim;      // 16,777,216
  const int nW = 2 * H3 * Hdim;          // 1,572,864 per weight tensor

  k_cvt_bf16<<<(nX + 255) / 256, 256, 0, stream>>>(x, xbf, nX);
  k_cvt_bf16<<<(nW + 255) / 256, 256, 0, stream>>>(Wi, wi0, nW);
  k_cvt_bf16<<<(nW + 255) / 256, 256, 0, stream>>>(Wh, wh0, nW);
  k_init<<<128, 256, 0, stream>>>(h0, hf, hbf, ctrl);
  k_xgemm<<<(Bsz * Tlen) / 16, 256, 0, stream>>>(xbf, wi0, xg0);
  k_gru<<<NWG, 128, 0, stream>>>(xg0, wi1, wh0, wh1, gr, br, gz, bz, gn, bn,
                                 hf, hbf, sRZ, sN, ctrl, out, hn_out);
}
